// BHViTSelfAttention_50672024158257
// MI455X (gfx1250) — matrix operand order repaired and checked by reading, not compiled, not run on hardware
//
#include <hip/hip_runtime.h>
#include <stdint.h>

// ---- problem constants ----
#define CCH   384
#define TPW   113          // tokens per window (49 + 64)
#define NWIN  512          // B * 64 windows
#define TT    (NWIN*TPW)   // 57856 total tokens (divisible by 128)
#define NHD   12
#define HDD   32
#define BB    8
#define HH    56

typedef __attribute__((ext_vector_type(16))) _Float16 v16h;
typedef __attribute__((ext_vector_type(8)))  float    v8f;

union Frag { v16h h; uint32_t u[8]; };

__device__ __forceinline__ v8f wmma_f16(v16h a, v16h b, v8f c) {
  return __builtin_amdgcn_wmma_f32_16x16x32_f16(false, a, false, b, (short)0, c, false, false);
}

// A-matrix 16x32 f16 lane/VGPR K pattern (ISA 7.12.2)
__device__ __forceinline__ int pkA(int v, int hi) {
  return (v < 4) ? (hi*8 + 2*v) : (16 + hi*8 + 2*(v-4));
}
// B-matrix 32x16 f16 K pattern (same as A: lanes 0-15: K=0..7,16..23, lanes 16-31: K=8..15,24..31)
__device__ __forceinline__ int pkB(int v, int hi) {
  return (v < 4) ? (hi*8 + 2*v) : (16 + hi*8 + 2*(v-4));
}

// ---------------------------------------------------------------------------
// K0: binarize weights  wb = sign(W)*mean|W|row  (f16), bias' = b + move@wb^T
// ---------------------------------------------------------------------------
__global__ void __launch_bounds__(128)
k_binarize(const float* Wq, const float* Wk, const float* Wv,
           const float* bq, const float* bk, const float* bv,
           const float* mvq, const float* mvk, const float* mvv,
           _Float16* wb, float* biasw) {
  int row = blockIdx.x % CCH;
  int m   = blockIdx.x / CCH;
  const float* W  = (m == 0) ? Wq  : (m == 1) ? Wk  : Wv;
  const float* bI = (m == 0) ? bq  : (m == 1) ? bk  : bv;
  const float* mv = (m == 0) ? mvq : (m == 1) ? mvk : mvv;
  int tid = threadIdx.x;
  __shared__ float red[128];
  float s = 0.f;
  for (int c = tid; c < CCH; c += 128) s += fabsf(W[(size_t)row*CCH + c]);
  red[tid] = s; __syncthreads();
  for (int off = 64; off > 0; off >>= 1) { if (tid < off) red[tid] += red[tid+off]; __syncthreads(); }
  float scale = red[0] / (float)CCH;
  __syncthreads();
  float bs = 0.f;
  for (int c = tid; c < CCH; c += 128) {
    float wv = W[(size_t)row*CCH + c];
    float q  = (wv > 0.f) ? scale : (wv < 0.f ? -scale : 0.f);
    wb[((size_t)m*CCH + row)*CCH + c] = (_Float16)q;
    bs += mv[c] * q;
  }
  red[tid] = bs; __syncthreads();
  for (int off = 64; off > 0; off >>= 1) { if (tid < off) red[tid] += red[tid+off]; __syncthreads(); }
  if (tid == 0) biasw[m*CCH + row] = bI[row] + red[0];
}

// ---------------------------------------------------------------------------
// K1: 7x7 stride-7 pooling (pad 3) -> merge = a1*max + (1-a1)*avg   (B,64,C)
// ---------------------------------------------------------------------------
__global__ void k_pool_merge(const float* __restrict__ x, const float* __restrict__ a1,
                             float* __restrict__ merge) {
  int idx = blockIdx.x * blockDim.x + threadIdx.x;
  if (idx >= BB*64*CCH) return;
  int c = idx % CCH;
  int p = (idx / CCH) & 63;
  int b = idx / (CCH*64);
  int oy = p >> 3, ox = p & 7;
  const float* xc = x + ((size_t)(b*CCH + c))*HH*HH;
  float sum = 0.f, mx = -3.4e38f;
  for (int dy = 0; dy < 7; ++dy) {
    int y = oy*7 - 3 + dy;
    if (y < 0 || y >= HH) continue;
    for (int dx = 0; dx < 7; ++dx) {
      int xx = ox*7 - 3 + dx;
      if (xx < 0 || xx >= HH) continue;
      float v = xc[y*HH + xx];
      sum += v; mx = fmaxf(mx, v);
    }
  }
  float avg = sum * (1.f/49.f);
  float al  = a1[c];
  merge[idx] = al*mx + (1.f - al)*avg;
}

// ---------------------------------------------------------------------------
// K2: gather tokens + LayerNorm -> tok (f32) and tokh (f16)
// ---------------------------------------------------------------------------
__global__ void __launch_bounds__(128)
k_tok(const float* __restrict__ x, const float* __restrict__ merge,
      const float* __restrict__ g, const float* __restrict__ bta,
      float* __restrict__ tok, _Float16* __restrict__ tokh) {
  int token = blockIdx.x;            // 0..TT-1
  int n = token / TPW, t = token % TPW;
  int b = n >> 6, wy = (n >> 3) & 7, wx = n & 7;
  int tid = threadIdx.x;
  __shared__ float r1[128], r2[128];
  float vals[3];
  #pragma unroll
  for (int i = 0; i < 3; ++i) {
    int c = tid + i*128;
    float v;
    if (t < 49) {
      int iy = t / 7, ix = t % 7;
      int y = wy*7 + iy, xx = wx*7 + ix;
      v = x[(((size_t)b*CCH + c)*HH + y)*HH + xx];
    } else {
      v = merge[((size_t)b*64 + (t - 49))*CCH + c];
    }
    vals[i] = v;
  }
  float s = vals[0]+vals[1]+vals[2];
  float q = vals[0]*vals[0]+vals[1]*vals[1]+vals[2]*vals[2];
  r1[tid] = s; r2[tid] = q; __syncthreads();
  for (int off = 64; off > 0; off >>= 1) {
    if (tid < off) { r1[tid] += r1[tid+off]; r2[tid] += r2[tid+off]; }
    __syncthreads();
  }
  float mean = r1[0] / (float)CCH;
  float var  = r2[0] / (float)CCH - mean*mean;
  float rs   = rsqrtf(var + 1e-12f);
  size_t base = (size_t)token * CCH;
  #pragma unroll
  for (int i = 0; i < 3; ++i) {
    int c = tid + i*128;
    float o = (vals[i] - mean) * rs * g[c] + bta[c];
    tok[base + c]  = o;
    tokh[base + c] = (_Float16)o;
  }
}

// ---------------------------------------------------------------------------
// K3: WMMA GEMM  Y[TT x 384] = tokh[TT x 384] * Wb^T[384 x 384]
//     block: 256 threads = 8 waves; tile 128(M) x 64(N); weights in LDS
// ---------------------------------------------------------------------------
__global__ void __launch_bounds__(256)
k_gemm(const _Float16* __restrict__ A, const _Float16* __restrict__ Wb,
       float* __restrict__ Y) {
  __shared__ _Float16 wt[64 * CCH];     // 48 KB
  int tid  = threadIdx.x;
  int mBase = blockIdx.x * 128;
  int nBase = blockIdx.y * 64;
  // stage 64 weight rows (full K) into LDS as dwords
  const uint32_t* wg  = (const uint32_t*)(Wb + (size_t)nBase * CCH);
  uint32_t*       wls = (uint32_t*)wt;
  for (int i = tid; i < 64 * (CCH/2); i += 256) wls[i] = wg[i];
  __syncthreads();

  int lane = tid & 31, wave = tid >> 5;
  int hi = lane >> 4, ln = lane & 15;
  int wm = (wave & 3) * 32, wn = (wave >> 2) * 32;
  int pa[8], pb[8];
  #pragma unroll
  for (int v = 0; v < 8; ++v) { pa[v] = pkA(v, hi); pb[v] = pkB(v, hi); }

  v8f acc[2][2] = {};
  for (int k0 = 0; k0 < CCH; k0 += 32) {
    Frag a[2], bf[2];
    #pragma unroll
    for (int i = 0; i < 2; ++i) {
      const _Float16* ap = A + ((size_t)(mBase + wm + i*16 + ln))*CCH + k0;
      #pragma unroll
      for (int v = 0; v < 8; ++v) a[i].u[v] = *(const uint32_t*)(ap + pa[v]);
    }
    #pragma unroll
    for (int j = 0; j < 2; ++j) {
      const _Float16* bp = wt + (size_t)(wn + j*16 + ln)*CCH + k0;
      #pragma unroll
      for (int v = 0; v < 8; ++v) bf[j].u[v] = *(const uint32_t*)(bp + pb[v]);
    }
    #pragma unroll
    for (int i = 0; i < 2; ++i)
      #pragma unroll
      for (int j = 0; j < 2; ++j)
        acc[i][j] = wmma_f16(a[i].h, bf[j].h, acc[i][j]);
  }
  #pragma unroll
  for (int i = 0; i < 2; ++i)
    #pragma unroll
    for (int j = 0; j < 2; ++j) {
      int col = nBase + wn + j*16 + ln;
      #pragma unroll
      for (int v = 0; v < 8; ++v) {
        int row = mBase + wm + i*16 + v + hi*8;
        Y[(size_t)row*CCH + col] = acc[i][j][v];
      }
    }
}

// ---------------------------------------------------------------------------
// K4: per-branch  m = rprelu(LN(Y+bias') + tok);  msum (+)= m;  outh = f16(m+move2)
// ---------------------------------------------------------------------------
__global__ void __launch_bounds__(128)
k_lnres(const float* __restrict__ Y, const float* __restrict__ biasw,
        const float* __restrict__ tok,
        const float* __restrict__ g, const float* __restrict__ bb,
        const float* __restrict__ m1, const float* __restrict__ aa, const float* __restrict__ m2,
        const float* __restrict__ move2,
        float* __restrict__ msum, _Float16* __restrict__ outh, int first) {
  int token = blockIdx.x, tid = threadIdx.x;
  __shared__ float r1[128], r2[128];
  size_t base = (size_t)token * CCH;
  float vals[3];
  #pragma unroll
  for (int i = 0; i < 3; ++i) { int c = tid + i*128; vals[i] = Y[base + c] + biasw[c]; }
  float s = vals[0]+vals[1]+vals[2];
  float q = vals[0]*vals[0]+vals[1]*vals[1]+vals[2]*vals[2];
  r1[tid] = s; r2[tid] = q; __syncthreads();
  for (int off = 64; off > 0; off >>= 1) {
    if (tid < off) { r1[tid] += r1[tid+off]; r2[tid] += r2[tid+off]; }
    __syncthreads();
  }
  float mean = r1[0] / (float)CCH;
  float var  = r2[0] / (float)CCH - mean*mean;
  float rs   = rsqrtf(var + 1e-12f);
  #pragma unroll
  for (int i = 0; i < 3; ++i) {
    int c = tid + i*128;
    float o = (vals[i] - mean)*rs*g[c] + bb[c] + tok[base + c];
    float y = o - m1[c];
    float r = (y > 0.f ? y : aa[c]*y) + m2[c];
    if (first) msum[base + c] = r; else msum[base + c] += r;
    outh[base + c] = (_Float16)(r + move2[c]);
  }
}

// ---------------------------------------------------------------------------
// K5: attention per (window, head). 256 threads = 8 waves; dynamic LDS:
//     S (128x128 f32) + P (128x128 f16) = 96 KB
// ---------------------------------------------------------------------------
__global__ void __launch_bounds__(256)
k_attn(const _Float16* __restrict__ qf, const _Float16* __restrict__ kf,
       const _Float16* __restrict__ vf, float* __restrict__ ctx) {
  extern __shared__ char smem[];
  float*    S = (float*)smem;                       // 128*128 f32
  _Float16* P = (_Float16*)(smem + 128*128*4);      // 128*128 f16
  int n = blockIdx.x, h = blockIdx.y;
  int tid = threadIdx.x, lane = tid & 31, wave = tid >> 5;
  int hi = lane >> 4, ln = lane & 15;
  int pa[8], pb[8];
  #pragma unroll
  for (int v = 0; v < 8; ++v) { pa[v] = pkA(v, hi); pb[v] = pkB(v, hi); }
  size_t qbase = (size_t)n * TPW * CCH + h * HDD;
  const float scale = 0.17677669529663687f;   // 1/sqrt(32)

  // ---- S = Q K^T / sqrt(d) ----
  Frag a;
  int ta = wave*16 + ln;
  if (ta < TPW) {
    const _Float16* ap = qf + qbase + (size_t)ta * CCH;
    #pragma unroll
    for (int v = 0; v < 8; ++v) a.u[v] = *(const uint32_t*)(ap + pa[v]);
  } else {
    #pragma unroll
    for (int v = 0; v < 8; ++v) a.u[v] = 0u;
  }
  for (int j = 0; j < 8; ++j) {
    Frag bf;
    int tb = j*16 + ln;
    if (tb < TPW) {
      const _Float16* bp = kf + qbase + (size_t)tb * CCH;
      #pragma unroll
      for (int v = 0; v < 8; ++v) bf.u[v] = *(const uint32_t*)(bp + pb[v]);
    } else {
      #pragma unroll
      for (int v = 0; v < 8; ++v) bf.u[v] = 0u;
    }
    v8f acc = {};
    acc = wmma_f16(a.h, bf.h, acc);
    int col = j*16 + ln;
    #pragma unroll
    for (int v = 0; v < 8; ++v) {
      int row = wave*16 + v + hi*8;
      float val = acc[v] * scale;
      if (col >= TPW) val = -1.0e30f;
      S[row*128 + col] = val;
    }
  }
  __syncthreads();

  // ---- softmax rows -> P (f16) ----
  if (tid < 128) {
    int r = tid;
    float mx = -3.4e38f;
    for (int c = 0; c < TPW; ++c) mx = fmaxf(mx, S[r*128 + c]);
    float sum = 0.f;
    for (int c = 0; c < TPW; ++c) sum += __expf(S[r*128 + c] - mx);
    float inv = 1.f / sum;
    for (int c = 0; c < 128; ++c)
      P[r*128 + c] = (c < TPW) ? (_Float16)(__expf(S[r*128 + c] - mx) * inv) : (_Float16)0.f;
  }
  __syncthreads();

  // ---- ctx = P V ----
  v8f acc2[2] = {};
  const uint32_t* Pu = (const uint32_t*)P;
  const unsigned short* vfu = (const unsigned short*)vf;
  int rowA = wave*16 + ln;
  for (int kk = 0; kk < 4; ++kk) {
    Frag ap;
    #pragma unroll
    for (int v = 0; v < 8; ++v) ap.u[v] = Pu[rowA*64 + ((kk*32 + pa[v]) >> 1)];
    #pragma unroll
    for (int j2 = 0; j2 < 2; ++j2) {
      Frag bf;
      int dcol = j2*16 + ln;
      #pragma unroll
      for (int v = 0; v < 8; ++v) {
        int tk = kk*32 + pb[v];
        unsigned int lo  = (tk     < TPW) ? (unsigned int)vfu[qbase + (size_t)tk     * CCH + dcol] : 0u;
        unsigned int hi2 = (tk + 1 < TPW) ? (unsigned int)vfu[qbase + (size_t)(tk+1) * CCH + dcol] : 0u;
        bf.u[v] = lo | (hi2 << 16);
      }
      acc2[j2] = wmma_f16(ap.h, bf.h, acc2[j2]);
    }
  }
  #pragma unroll
  for (int j2 = 0; j2 < 2; ++j2) {
    int d = j2*16 + ln;
    #pragma unroll
    for (int v = 0; v < 8; ++v) {
      int t = wave*16 + v + hi*8;
      if (t < TPW) ctx[qbase + (size_t)t * CCH + d] = acc2[j2][v];
    }
  }
}

// ---------------------------------------------------------------------------
// K6a: ctx = rprelu(LN(ctx) + msum)   (in place, one block per token)
// ---------------------------------------------------------------------------
__global__ void __launch_bounds__(128)
k_ctxln(float* __restrict__ ctx, const float* __restrict__ msum,
        const float* __restrict__ g, const float* __restrict__ bb,
        const float* __restrict__ m1, const float* __restrict__ aa, const float* __restrict__ m2) {
  int token = blockIdx.x, tid = threadIdx.x;
  __shared__ float r1[128], r2[128];
  size_t base = (size_t)token * CCH;
  float vals[3];
  #pragma unroll
  for (int i = 0; i < 3; ++i) { int c = tid + i*128; vals[i] = ctx[base + c]; }
  float s = vals[0]+vals[1]+vals[2];
  float q = vals[0]*vals[0]+vals[1]*vals[1]+vals[2]*vals[2];
  r1[tid] = s; r2[tid] = q; __syncthreads();
  for (int off = 64; off > 0; off >>= 1) {
    if (tid < off) { r1[tid] += r1[tid+off]; r2[tid] += r2[tid+off]; }
    __syncthreads();
  }
  float mean = r1[0] / (float)CCH;
  float var  = r2[0] / (float)CCH - mean*mean;
  float rs   = rsqrtf(var + 1e-12f);
  #pragma unroll
  for (int i = 0; i < 3; ++i) {
    int c = tid + i*128;
    float o = (vals[i] - mean)*rs*g[c] + bb[c] + msum[base + c];
    float y = o - m1[c];
    ctx[base + c] = (y > 0.f ? y : aa[c]*y) + m2[c];
  }
}

// ---------------------------------------------------------------------------
// K6b: x2[b,p,c] = mean over 64 windows of ctx pooled tokens
// ---------------------------------------------------------------------------
__global__ void k_x2(const float* __restrict__ ctx, float* __restrict__ x2b) {
  int idx = blockIdx.x * blockDim.x + threadIdx.x;
  if (idx >= BB*64*CCH) return;
  int c = idx % CCH;
  int p = (idx / CCH) & 63;
  int b = idx / (CCH*64);
  float s = 0.f;
  for (int w = 0; w < 64; ++w)
    s += ctx[((size_t)(b*64 + w)*TPW + 49 + p)*CCH + c];
  x2b[idx] = s * (1.f/64.f);
}

// ---------------------------------------------------------------------------
// K6c: out = x1*parm + upsample(x2)*(1-parm)
// ---------------------------------------------------------------------------
__global__ void k_out(const float* __restrict__ ctx, const float* __restrict__ x2b,
                      const float* __restrict__ parm, float* __restrict__ out, int total) {
  int idx = blockIdx.x * blockDim.x + threadIdx.x;
  if (idx >= total) return;
  int xx = idx % HH;
  int y  = (idx / HH) % HH;
  int c  = (idx / (HH*HH)) % CCH;
  int b  = idx / (HH*HH*CCH);
  int wy = y / 7, iy = y % 7, wx = xx / 7, ix = xx % 7;
  int n = (b << 6) + wy*8 + wx;
  int t = iy*7 + ix;
  float x1v = ctx[((size_t)n*TPW + t)*CCH + c];
  float x2v = x2b[((size_t)b*64 + wy*8 + wx)*CCH + c];
  float p = parm[c];
  out[idx] = x1v*p + x2v*(1.f - p);
}

// ---------------------------------------------------------------------------
extern "C" void kernel_launch(void* const* d_in, const int* in_sizes, int n_in,
                              void* d_out, int out_size, void* d_ws, size_t ws_size,
                              hipStream_t stream) {
  (void)in_sizes; (void)n_in; (void)out_size; (void)ws_size;
  const float* x        = (const float*)d_in[0];
  const float* a1       = (const float*)d_in[1];
  const float* ln_tok_g = (const float*)d_in[2];
  const float* ln_tok_b = (const float*)d_in[3];
  const float* moveq    = (const float*)d_in[4];
  const float* movek    = (const float*)d_in[5];
  const float* movev    = (const float*)d_in[6];
  const float* Wq       = (const float*)d_in[7];
  const float* bq       = (const float*)d_in[8];
  const float* Wk       = (const float*)d_in[9];
  const float* bk       = (const float*)d_in[10];
  const float* Wv       = (const float*)d_in[11];
  const float* bv       = (const float*)d_in[12];
  const float* lng[3]   = {(const float*)d_in[13], (const float*)d_in[15], (const float*)d_in[17]};
  const float* lnb[3]   = {(const float*)d_in[14], (const float*)d_in[16], (const float*)d_in[18]};
  const float* rm1[3]   = {(const float*)d_in[19], (const float*)d_in[22], (const float*)d_in[25]};
  const float* ra[3]    = {(const float*)d_in[20], (const float*)d_in[23], (const float*)d_in[26]};
  const float* rm2[3]   = {(const float*)d_in[21], (const float*)d_in[24], (const float*)d_in[27]};
  const float* mv2[3]   = {(const float*)d_in[28], (const float*)d_in[29], (const float*)d_in[30]};
  const float* lnc_g    = (const float*)d_in[31];
  const float* lnc_b    = (const float*)d_in[32];
  const float* rpc_m1   = (const float*)d_in[33];
  const float* rpc_a    = (const float*)d_in[34];
  const float* rpc_m2   = (const float*)d_in[35];
  const float* parm     = (const float*)d_in[36];
  float* out = (float*)d_out;

  char* wptr = (char*)d_ws;
  auto alloc = [&](size_t bytes) -> char* {
    char* p = wptr; wptr += (bytes + 255) & ~(size_t)255; return p;
  };
  float*    tok   = (float*)   alloc((size_t)TT*CCH*4);
  _Float16* tokh  = (_Float16*)alloc((size_t)TT*CCH*2);
  float*    Y     = (float*)   alloc((size_t)TT*CCH*4);
  float*    msum  = (float*)   alloc((size_t)TT*CCH*4);
  _Float16* qh    = (_Float16*)alloc((size_t)TT*CCH*2);
  _Float16* kh    = (_Float16*)alloc((size_t)TT*CCH*2);
  _Float16* vh    = (_Float16*)alloc((size_t)TT*CCH*2);
  float*    merge = (float*)   alloc((size_t)BB*64*CCH*4);
  float*    x2b   = (float*)   alloc((size_t)BB*64*CCH*4);
  _Float16* wb    = (_Float16*)alloc((size_t)3*CCH*CCH*2);
  float*    biasw = (float*)   alloc((size_t)3*CCH*4);
  float*    ctx   = Y;   // Y is dead after k_lnres of the last branch

  k_binarize<<<3*CCH, 128, 0, stream>>>(Wq, Wk, Wv, bq, bk, bv, moveq, movek, movev, wb, biasw);
  k_pool_merge<<<(BB*64*CCH + 255)/256, 256, 0, stream>>>(x, a1, merge);
  k_tok<<<TT, 128, 0, stream>>>(x, merge, ln_tok_g, ln_tok_b, tok, tokh);

  _Float16* outs[3] = {qh, kh, vh};
  for (int br = 0; br < 3; ++br) {
    k_gemm<<<dim3(TT/128, CCH/64), 256, 0, stream>>>(tokh, wb + (size_t)br*CCH*CCH, Y);
    k_lnres<<<TT, 128, 0, stream>>>(Y, biasw + br*CCH, tok,
                                    lng[br], lnb[br], rm1[br], ra[br], rm2[br],
                                    mv2[br], msum, outs[br], br == 0 ? 1 : 0);
  }

  k_attn<<<dim3(NWIN, NHD), 256, 128*128*4 + 128*128*2, stream>>>(qh, kh, vh, ctx);
  k_ctxln<<<TT, 128, 0, stream>>>(ctx, msum, lnc_g, lnc_b, rpc_m1, rpc_a, rpc_m2);
  k_x2<<<(BB*64*CCH + 255)/256, 256, 0, stream>>>(ctx, x2b);
  int total = BB*CCH*HH*HH;
  k_out<<<(total + 255)/256, 256, 0, stream>>>(ctx, x2b, parm, out, total);
}
